// GCNEncoder_27212912787788
// MI455X (gfx1250) — compile-verified, matrix operand order checked by reading, not run on hardware
//
#include <hip/hip_runtime.h>
#include <hip/hip_bf16.h>

typedef __attribute__((ext_vector_type(16))) __bf16 v16bf;
typedef __attribute__((ext_vector_type(8)))  __bf16 v8bf;
typedef __attribute__((ext_vector_type(8)))  float  v8f;

#define FDIM 512
#define MT_PER_BLOCK 8   // 8 waves (256 threads) per block, one 16x16 tile per wave

// ---------------------------------------------------------------------------
// One-shot: transpose W[k][n] (f32) -> Wt[n][k] (bf16). 512KB result, L2-resident.
// ---------------------------------------------------------------------------
__global__ void k_w_to_bf16t(const float* __restrict__ W, __bf16* __restrict__ Wt,
                             int total /* = FDIM*FDIM */) {
    int idx = blockIdx.x * blockDim.x + threadIdx.x;
    if (idx >= total) return;
    int k = idx >> 9;            // / FDIM   (coalesced read)
    int n = idx & (FDIM - 1);
    Wt[(size_t)n * FDIM + k] = (__bf16)W[idx];
}

// ---------------------------------------------------------------------------
// WMMA GEMM: H[M,512] = act(X[M,512]) @ W[512,512]
// A: f32 loaded as 4x b128, converted in-register (cvt_pk_bf16_f32).
// B: pre-converted/transposed bf16, 2x b128 per chunk, zero converts.
// One wave computes a 16x16 tile; K loop = 16 x wmma_f32_16x16x32_bf16.
// ---------------------------------------------------------------------------
template <bool RELU>
__global__ __launch_bounds__(256)
void gcn_gemm_wmma(const float* __restrict__ X, const __bf16* __restrict__ Wt,
                   float* __restrict__ H, int mTiles) {
    const int lane = threadIdx.x & 31;
    const int wave = threadIdx.x >> 5;
    const int mTile = blockIdx.x * MT_PER_BLOCK + wave;
    if (mTile >= mTiles) return;                 // wave-uniform guard (EXEC stays full)
    const int nTile = blockIdx.y;

    const int ml    = lane & 15;                 // A row / B col / C col within tile
    const int kBase = (lane & 16) ? 8 : 0;       // ISA 16-bit operand layout: hi lanes K+8

    const float*  __restrict__ Ap0 = X  + (size_t)(mTile * 16 + ml) * FDIM + kBase;
    const __bf16* __restrict__ Bp0 = Wt + (size_t)(nTile * 16 + ml) * FDIM + kBase;

    v8f acc = {};
    for (int k0 = 0; k0 < FDIM; k0 += 32) {
        float4 a0 = *(const float4*)(Ap0 + k0);          // K = k0+kBase+0..3
        float4 a1 = *(const float4*)(Ap0 + k0 + 4);      // K = k0+kBase+4..7
        float4 a2 = *(const float4*)(Ap0 + k0 + 16);     // K = k0+16+kBase+0..3
        float4 a3 = *(const float4*)(Ap0 + k0 + 20);     // K = k0+16+kBase+4..7
        v8bf  blo = *(const v8bf*)(Bp0 + k0);
        v8bf  bhi = *(const v8bf*)(Bp0 + k0 + 16);

        if (RELU) {
            a0.x = fmaxf(a0.x, 0.f); a0.y = fmaxf(a0.y, 0.f); a0.z = fmaxf(a0.z, 0.f); a0.w = fmaxf(a0.w, 0.f);
            a1.x = fmaxf(a1.x, 0.f); a1.y = fmaxf(a1.y, 0.f); a1.z = fmaxf(a1.z, 0.f); a1.w = fmaxf(a1.w, 0.f);
            a2.x = fmaxf(a2.x, 0.f); a2.y = fmaxf(a2.y, 0.f); a2.z = fmaxf(a2.z, 0.f); a2.w = fmaxf(a2.w, 0.f);
            a3.x = fmaxf(a3.x, 0.f); a3.y = fmaxf(a3.y, 0.f); a3.z = fmaxf(a3.z, 0.f); a3.w = fmaxf(a3.w, 0.f);
        }

        v16bf a, b;
        a[0]  = (__bf16)a0.x; a[1]  = (__bf16)a0.y; a[2]  = (__bf16)a0.z; a[3]  = (__bf16)a0.w;
        a[4]  = (__bf16)a1.x; a[5]  = (__bf16)a1.y; a[6]  = (__bf16)a1.z; a[7]  = (__bf16)a1.w;
        a[8]  = (__bf16)a2.x; a[9]  = (__bf16)a2.y; a[10] = (__bf16)a2.z; a[11] = (__bf16)a2.w;
        a[12] = (__bf16)a3.x; a[13] = (__bf16)a3.y; a[14] = (__bf16)a3.z; a[15] = (__bf16)a3.w;
#pragma unroll
        for (int i = 0; i < 8; ++i) { b[i] = blo[i]; b[8 + i] = bhi[i]; }

        acc = __builtin_amdgcn_wmma_f32_16x16x32_bf16(
            /*neg_a=*/false, a, /*neg_b=*/false, b,
            /*c_mod=*/(short)0, acc, /*reuse_a=*/false, /*reuse_b=*/false);
    }

    // C/D layout: VGPR r -> row r + (lane>=16 ? 8 : 0), col = lane%16
    const int mOff = (lane & 16) ? 8 : 0;
    float* __restrict__ Hout = H + (size_t)(mTile * 16 + mOff) * FDIM + nTile * 16 + ml;
#pragma unroll
    for (int r = 0; r < 8; ++r)
        Hout[(size_t)r * FDIM] = acc[r];
}

// ---------------------------------------------------------------------------
// Degree / normalization helpers
// ---------------------------------------------------------------------------
__global__ void k_init_deg(float* __restrict__ deg, int n) {
    int i = blockIdx.x * blockDim.x + threadIdx.x;
    if (i < n) deg[i] = 1.0f;                    // self loop contributes 1
}

__global__ void k_deg_accum(const int* __restrict__ col, float* __restrict__ deg, int e) {
    int i = blockIdx.x * blockDim.x + threadIdx.x;
    if (i < e) atomicAdd(&deg[col[i]], 1.0f);
}

__global__ void k_deg_to_dis(float* __restrict__ deg, int n) {
    int i = blockIdx.x * blockDim.x + threadIdx.x;
    if (i < n) deg[i] = rsqrtf(deg[i]);          // deg >= 1 always
}

// out[i,f] = bias[f] + dis[i]^2 * h[i,f]  (bias + self-loop; full overwrite, float4)
__global__ void k_agg_init(const float* __restrict__ h, const float* __restrict__ dis,
                           const float* __restrict__ bias, float* __restrict__ out,
                           int total4 /* = N*FDIM/4 */) {
    int idx = blockIdx.x * blockDim.x + threadIdx.x;
    if (idx >= total4) return;
    int i  = idx >> 7;                 // / (FDIM/4)
    int f4 = (idx & 127) << 2;
    float d2 = dis[i]; d2 *= d2;
    float4 hv = *(const float4*)(h + (size_t)idx * 4);
    float4 bv = *(const float4*)(bias + f4);
    float4 ov;
    ov.x = bv.x + d2 * hv.x; ov.y = bv.y + d2 * hv.y;
    ov.z = bv.z + d2 * hv.z; ov.w = bv.w + d2 * hv.w;
    *(float4*)(out + (size_t)idx * 4) = ov;
}

// per-edge scatter: out[col] += h[row] * dis[row]*dis[col]
// 128 threads x float4 = one 512-feature row per block; gather reads are b128.
__global__ __launch_bounds__(128)
void k_edge_scatter(const float* __restrict__ h, const int* __restrict__ row,
                    const int* __restrict__ col, const float* __restrict__ dis,
                    float* __restrict__ out, int e) {
    int ei = blockIdx.x;
    if (ei >= e) return;
    int r = row[ei];
    int c = col[ei];
    float nrm = dis[r] * dis[c];
    const float* __restrict__ src = h   + (size_t)r * FDIM + threadIdx.x * 4;
    float*       __restrict__ dst = out + (size_t)c * FDIM + threadIdx.x * 4;
    float4 sv = *(const float4*)src;
    atomicAdd(dst + 0, sv.x * nrm);
    atomicAdd(dst + 1, sv.y * nrm);
    atomicAdd(dst + 2, sv.z * nrm);
    atomicAdd(dst + 3, sv.w * nrm);
}

// ---------------------------------------------------------------------------
// Launch: convert weights + deg/dis -> [GEMM -> agg_init -> scatter] x 3
// ---------------------------------------------------------------------------
extern "C" void kernel_launch(void* const* d_in, const int* in_sizes, int n_in,
                              void* d_out, int out_size, void* d_ws, size_t ws_size,
                              hipStream_t stream) {
    const float* x  = (const float*)d_in[0];
    const int*   ei = (const int*)d_in[1];        // [2,E] flat: row = ei[e], col = ei[E+e]
    const float* W1 = (const float*)d_in[2];
    const float* b1 = (const float*)d_in[3];
    const float* W2 = (const float*)d_in[4];
    const float* b2 = (const float*)d_in[5];
    const float* W3 = (const float*)d_in[6];
    const float* b3 = (const float*)d_in[7];

    const int N = in_sizes[0] / FDIM;             // 10000
    const int E = in_sizes[1] / 2;                // 160000
    const int* erow = ei;
    const int* ecol = ei + E;

    float*  h_buf   = (float*)d_ws;                             // [N,512] f32
    float*  agg_buf = h_buf + (size_t)N * FDIM;                 // [N,512] f32
    float*  dis     = agg_buf + (size_t)N * FDIM;               // [N] f32
    __bf16* wt1     = (__bf16*)(dis + N);                       // [512,512] bf16 (transposed)
    __bf16* wt2     = wt1 + (size_t)FDIM * FDIM;
    __bf16* wt3     = wt2 + (size_t)FDIM * FDIM;
    float*  outp    = (float*)d_out;

    const int mTiles = (N + 15) / 16;             // 625 (exact)
    dim3 gGemm((mTiles + MT_PER_BLOCK - 1) / MT_PER_BLOCK, FDIM / 16);
    const int wTotal  = FDIM * FDIM;
    const int wBlk    = (wTotal + 255) / 256;
    const int total4  = N * FDIM / 4;
    const int nBlk1   = (N + 255) / 256;
    const int eBlk1   = (E + 255) / 256;
    const int nf4Blk  = (total4 + 255) / 256;

    // --- weight transpose + bf16 convert (tiny, one-shot per call) ---
    k_w_to_bf16t<<<wBlk, 256, 0, stream>>>(W1, wt1, wTotal);
    k_w_to_bf16t<<<wBlk, 256, 0, stream>>>(W2, wt2, wTotal);
    k_w_to_bf16t<<<wBlk, 256, 0, stream>>>(W3, wt3, wTotal);

    // --- normalization: deg -> dis ---
    k_init_deg<<<nBlk1, 256, 0, stream>>>(dis, N);
    k_deg_accum<<<eBlk1, 256, 0, stream>>>(ecol, dis, E);
    k_deg_to_dis<<<nBlk1, 256, 0, stream>>>(dis, N);

    // --- layer 1: x -> h_buf -> agg_buf ---
    gcn_gemm_wmma<false><<<gGemm, 256, 0, stream>>>(x, wt1, h_buf, mTiles);
    k_agg_init<<<nf4Blk, 256, 0, stream>>>(h_buf, dis, b1, agg_buf, total4);
    k_edge_scatter<<<E, 128, 0, stream>>>(h_buf, erow, ecol, dis, agg_buf, E);

    // --- layer 2: relu(agg_buf) -> h_buf -> agg_buf ---
    gcn_gemm_wmma<true><<<gGemm, 256, 0, stream>>>(agg_buf, wt2, h_buf, mTiles);
    k_agg_init<<<nf4Blk, 256, 0, stream>>>(h_buf, dis, b2, agg_buf, total4);
    k_edge_scatter<<<E, 128, 0, stream>>>(h_buf, erow, ecol, dis, agg_buf, E);

    // --- layer 3: relu(agg_buf) -> h_buf -> d_out ---
    gcn_gemm_wmma<true><<<gGemm, 256, 0, stream>>>(agg_buf, wt3, h_buf, mTiles);
    k_agg_init<<<nf4Blk, 256, 0, stream>>>(h_buf, dis, b3, outp, total4);
    k_edge_scatter<<<E, 128, 0, stream>>>(h_buf, erow, ecol, dis, outp, E);
}
